// MessageProp_51376398795078
// MI455X (gfx1250) — matrix operand order repaired and checked by reading, not compiled, not run on hardware
//
#include <hip/hip_runtime.h>

// ---------------- problem constants ----------------
#define DD   128      // latent dim
#define BB   65536    // batch
#define KK   8        // components
#define HPN  144      // hidden out-dim padded (9 x 16 tiles, real 132)
#define HPK  160      // hidden in-dim padded  (5 x 32 k-tiles, real 132)

// ---------------- workspace / LDS layout (bytes) ----------------
#define OFF_WC 0                    // 128x128 bf16  = 32768
#define OFF_W0 32768                // 144x256 bf16  = 73728
#define OFF_W1 106496               // 144x160 bf16  = 46080
#define OFF_W2 152576               // 144x160 bf16  = 46080
#define OFF_W3 198656               // 128x160 bf16  = 40960
#define OFF_BC 239616               // 128 f32
#define OFF_B0 240128               // 144 f32
#define OFF_B1 240704               // 144 f32
#define OFF_B2 241280               // 144 f32
#define OFF_B3 241856               // 128 f32
#define WTOTAL 242368

#define ACT_LDK        160                       // activation strip width (bf16 elems)
#define ACT_WAVE_BYTES (16 * ACT_LDK * 2)        // 5120 per wave
#define ACT_OFF        WTOTAL
#define ACT_BYTES      (8 * ACT_WAVE_BYTES)      // 40960
#define SMEM_TOTAL     (WTOTAL + ACT_BYTES)      // 283328 < 320KB WGP LDS

typedef __attribute__((ext_vector_type(16))) __bf16 bf16x16;
typedef __attribute__((ext_vector_type(8)))  __bf16 bf16x8;
typedef __attribute__((ext_vector_type(8)))  float  f32x8;
typedef __attribute__((ext_vector_type(4)))  float  f32x4;
typedef __attribute__((ext_vector_type(4)))  unsigned int u32x4;

static __device__ __forceinline__ f32x8 wmma_bf16(bf16x16 a, bf16x16 b, f32x8 c) {
    return __builtin_amdgcn_wmma_f32_16x16x32_bf16(
        /*neg_a=*/false, a, /*neg_b=*/false, b,
        /*c_mod=*/(short)0, c, /*reuse_a=*/false, /*reuse_b=*/false);
}

// B-fragment (32x16, KxN): lane n=lane&15 reads W[n0+n][k0 + (lane>>4)*8 .. +7] and [.. +16 .. +23]
// (W stored row-major [out,in] bf16; two 16B chunks per lane)
static __device__ __forceinline__ bf16x16 load_b_frag(const __bf16* W, int ldk,
                                                      int n0, int k0, int lane) {
    const __bf16* p = W + (size_t)(n0 + (lane & 15)) * ldk + k0 + ((lane >> 4) << 3);
    union { bf16x16 v; bf16x8 h[2]; } u;
    u.h[0] = *(const bf16x8*)(p);
    u.h[1] = *(const bf16x8*)(p + 16);
    return u.v;
}

// A-fragment (16x32) from LDS bf16 row-major act strip:
// elems 0..7 = K[k0+s8 .. +7], elems 8..15 = K[k0+16+s8 .. +7], row m=lane&15
static __device__ __forceinline__ bf16x16 load_a_frag_lds(const __bf16* A, int ldk,
                                                          int k0, int lane) {
    int m = lane & 15, s8 = (lane >> 4) * 8;
    const __bf16* p = A + m * ldk + k0 + s8;
    union { bf16x16 v; bf16x8 h[2]; } u;
    u.h[0] = *(const bf16x8*)(p);
    u.h[1] = *(const bf16x8*)(p + 16);
    return u.v;
}

// A-fragment from global f32 matrix with on-the-fly bf16 convert
static __device__ __forceinline__ bf16x16 load_a_frag_g32(const float* A, int ld,
                                                          int row0, int k0, int lane) {
    int m = lane & 15, s8 = (lane >> 4) * 8;
    const float* p = A + (size_t)(row0 + m) * ld + k0 + s8;
    f32x4 a0 = *(const f32x4*)(p);
    f32x4 a1 = *(const f32x4*)(p + 4);
    f32x4 a2 = *(const f32x4*)(p + 16);
    f32x4 a3 = *(const f32x4*)(p + 20);
    bf16x16 r;
#pragma unroll
    for (int j = 0; j < 4; ++j) {
        r[j]      = (__bf16)a0[j];
        r[4 + j]  = (__bf16)a1[j];
        r[8 + j]  = (__bf16)a2[j];
        r[12 + j] = (__bf16)a3[j];
    }
    return r;
}

// Software-pipelined layer: B-fragment for step i+1 is issued before the WMMA of
// step i (pipelined across k-tiles AND output tiles) so ds_load latency overlaps
// the WMMA chain instead of stalling on s_wait_dscnt 0.
template<int NK, int NT, typename EMIT>
static __device__ __forceinline__ void layer_mm(const bf16x16* aF, const __bf16* W,
                                                int ldk, const float* bias,
                                                int lane, int m, EMIT emit) {
    bf16x16 b = load_b_frag(W, ldk, 0, 0, lane);
    f32x8 c;
#pragma unroll
    for (int v = 0; v < 8; ++v) c[v] = 0.f;
#pragma unroll
    for (int ot = 0; ot < NT; ++ot) {
#pragma unroll
        for (int kt = 0; kt < NK; ++kt) {
            bf16x16 bn = b;
            if (!(ot == NT - 1 && kt == NK - 1)) {
                int kt2 = kt + 1, ot2 = ot;
                if (kt2 == NK) { kt2 = 0; ot2 = ot + 1; }
                bn = load_b_frag(W, ldk, ot2 * 16, kt2 * 32, lane);
            }
            c = wmma_bf16(aF[kt], b, c);
            b = bn;
        }
        emit(ot, c, bias[ot * 16 + m]);
#pragma unroll
        for (int v = 0; v < 8; ++v) c[v] = 0.f;
    }
}

// ---------------- prep kernels ----------------
// Wc = Wu @ Wm (bf16), bc = Wu @ (K*bm) + bu
__global__ void prep_wc(const float* __restrict__ Wm, const float* __restrict__ bm,
                        const float* __restrict__ Wu, const float* __restrict__ bu,
                        __bf16* __restrict__ wc, float* __restrict__ bc) {
    int i = blockIdx.x;   // out row 0..127
    int j = threadIdx.x;  // in col 0..127
    float s = 0.f;
    for (int t = 0; t < DD; ++t) s += Wu[i * DD + t] * Wm[t * DD + j];
    wc[i * DD + j] = (__bf16)s;
    if (j == 0) {
        float sb = bu[i];
        for (int t = 0; t < DD; ++t) sb += Wu[i * DD + t] * ((float)KK * bm[t]);
        bc[i] = sb;
    }
}

// zero-padded f32 -> bf16 weight convert + padded f32 bias
__global__ void prep_pad(const float* __restrict__ W, const float* __restrict__ b,
                         __bf16* __restrict__ Wp, float* __restrict__ bp,
                         int outDim, int inDim, int outPad, int inPad) {
    int idx = blockIdx.x * blockDim.x + threadIdx.x;
    int total = outPad * inPad;
    if (idx < total) {
        int o = idx / inPad, i = idx % inPad;
        float v = (o < outDim && i < inDim) ? W[o * inDim + i] : 0.f;
        Wp[idx] = (__bf16)v;
    }
    if (idx < outPad) bp[idx] = (idx < outDim) ? b[idx] : 0.f;
}

// ---------------- fused main kernel ----------------
__global__ __launch_bounds__(256)
void mp_fused(const float* __restrict__ signal,
              const float* __restrict__ comps,
              const unsigned char* __restrict__ ws,
              float* __restrict__ out) {
    extern __shared__ __align__(16) unsigned char smem[];

    // cooperative: weights+biases -> LDS, zero activation strips
    {
        const u32x4* s = (const u32x4*)ws;
        u32x4* d = (u32x4*)smem;
        for (int i = threadIdx.x; i < WTOTAL / 16; i += 256) d[i] = s[i];
        u32x4 z; z[0] = 0u; z[1] = 0u; z[2] = 0u; z[3] = 0u;
        u32x4* a = (u32x4*)(smem + ACT_OFF);
        for (int i = threadIdx.x; i < ACT_BYTES / 16; i += 256) a[i] = z;
    }
    __syncthreads();

    const int lane = threadIdx.x & 31;
    const int wave = threadIdx.x >> 5;
    const int row0 = blockIdx.x * 128 + wave * 16;
    const int m    = lane & 15;
    const int s8   = (lane >> 4) * 8;   // also the C/D row offset for this half-wave

    const __bf16* sWc = (const __bf16*)(smem + OFF_WC);
    const __bf16* sW0 = (const __bf16*)(smem + OFF_W0);
    const __bf16* sW1 = (const __bf16*)(smem + OFF_W1);
    const __bf16* sW2 = (const __bf16*)(smem + OFF_W2);
    const __bf16* sW3 = (const __bf16*)(smem + OFF_W3);
    const float* sbc = (const float*)(smem + OFF_BC);
    const float* sb0 = (const float*)(smem + OFF_B0);
    const float* sb1 = (const float*)(smem + OFF_B1);
    const float* sb2 = (const float*)(smem + OFF_B2);
    const float* sb3 = (const float*)(smem + OFF_B3);
    __bf16* act = (__bf16*)(smem + ACT_OFF + wave * ACT_WAVE_BYTES);

    // ---- stage 1: S = sum_k components[k], accumulated straight into A-frag layout
    float acc[4][16];
#pragma unroll
    for (int kt = 0; kt < 4; ++kt)
#pragma unroll
        for (int e = 0; e < 16; ++e) acc[kt][e] = 0.f;

#pragma unroll
    for (int k = 0; k < KK; ++k) {
        const float* base = comps + ((size_t)k * BB + (row0 + m)) * DD;
        if (k + 1 < KK)
            __builtin_prefetch(comps + ((size_t)(k + 1) * BB + (row0 + m)) * DD, 0, 1);
#pragma unroll
        for (int kt = 0; kt < 4; ++kt) {
            const float* p = base + kt * 32 + s8;
            f32x4 a0 = *(const f32x4*)(p);
            f32x4 a1 = *(const f32x4*)(p + 4);
            f32x4 a2 = *(const f32x4*)(p + 16);
            f32x4 a3 = *(const f32x4*)(p + 20);
#pragma unroll
            for (int j = 0; j < 4; ++j) {
                acc[kt][j]      += a0[j];
                acc[kt][4 + j]  += a1[j];
                acc[kt][8 + j]  += a2[j];
                acc[kt][12 + j] += a3[j];
            }
        }
    }
    bf16x16 aS[4];
#pragma unroll
    for (int kt = 0; kt < 4; ++kt)
#pragma unroll
        for (int e = 0; e < 16; ++e) aS[kt][e] = (__bf16)acc[kt][e];

    // ---- stage 2: right = S @ Wc^T + bc  -> act[rows][0..127] (bf16), no relu
    layer_mm<4, 8>(aS, sWc, DD, sbc, lane, m,
        [&](int ot, f32x8 c, float bv) {
#pragma unroll
            for (int v = 0; v < 8; ++v)
                act[(v + s8) * ACT_LDK + ot * 16 + m] = (__bf16)(c[v] + bv);
        });

    // ---- stage 3 (L0): h = relu(concat(signal, right) @ W0^T + b0)
    bf16x16 aL[8];
#pragma unroll
    for (int kt = 0; kt < 4; ++kt)
        aL[kt] = load_a_frag_g32(signal, DD, row0, kt * 32, lane);     // cols 0..127
#pragma unroll
    for (int kt = 0; kt < 4; ++kt)
        aL[4 + kt] = load_a_frag_lds(act, ACT_LDK, kt * 32, lane);     // cols 128..255
    layer_mm<8, 9>(aL, sW0, 256, sb0, lane, m,
        [&](int ot, f32x8 c, float bv) {
#pragma unroll
            for (int v = 0; v < 8; ++v) {
                float e = c[v] + bv;
                e = e > 0.f ? e : 0.f;
                act[(v + s8) * ACT_LDK + ot * 16 + m] = (__bf16)e;
            }
        });

    // ---- stages 4,5 (L1, L2): h = relu(h @ W^T + b), padded K=160, N=144
    const __bf16* Wh[2] = { sW1, sW2 };
    const float*  bh[2] = { sb1, sb2 };
#pragma unroll
    for (int layer = 0; layer < 2; ++layer) {
        bf16x16 aH[5];
#pragma unroll
        for (int kt = 0; kt < 5; ++kt)
            aH[kt] = load_a_frag_lds(act, ACT_LDK, kt * 32, lane);
        layer_mm<5, 9>(aH, Wh[layer], HPK, bh[layer], lane, m,
            [&](int ot, f32x8 c, float bv) {
#pragma unroll
                for (int v = 0; v < 8; ++v) {
                    float e = c[v] + bv;
                    e = e > 0.f ? e : 0.f;
                    act[(v + s8) * ACT_LDK + ot * 16 + m] = (__bf16)e;
                }
            });
    }

    // ---- stage 6 (L3): out = h @ W3^T + b3, f32 D-fragments straight to global
    {
        bf16x16 aH[5];
#pragma unroll
        for (int kt = 0; kt < 5; ++kt)
            aH[kt] = load_a_frag_lds(act, ACT_LDK, kt * 32, lane);
        layer_mm<5, 8>(aH, sW3, HPK, sb3, lane, m,
            [&](int ot, f32x8 c, float bv) {
#pragma unroll
                for (int v = 0; v < 8; ++v)
                    out[(size_t)(row0 + v + s8) * DD + ot * 16 + m] = c[v] + bv;
            });
    }
}

// ---------------- launcher ----------------
extern "C" void kernel_launch(void* const* d_in, const int* in_sizes, int n_in,
                              void* d_out, int out_size, void* d_ws, size_t ws_size,
                              hipStream_t stream) {
    const float* signal = (const float*)d_in[0];
    const float* comps  = (const float*)d_in[1];
    const float* Wm = (const float*)d_in[2];
    const float* bm = (const float*)d_in[3];
    const float* Wu = (const float*)d_in[4];
    const float* bu = (const float*)d_in[5];
    const float* W0 = (const float*)d_in[6];
    const float* b0 = (const float*)d_in[7];
    const float* W1 = (const float*)d_in[8];
    const float* b1 = (const float*)d_in[9];
    const float* W2 = (const float*)d_in[10];
    const float* b2 = (const float*)d_in[11];
    const float* W3 = (const float*)d_in[12];
    const float* b3 = (const float*)d_in[13];
    unsigned char* ws = (unsigned char*)d_ws;

    prep_wc<<<DD, DD, 0, stream>>>(Wm, bm, Wu, bu,
                                   (__bf16*)(ws + OFF_WC), (float*)(ws + OFF_BC));
    prep_pad<<<(HPN * 256 + 255) / 256, 256, 0, stream>>>(
        W0, b0, (__bf16*)(ws + OFF_W0), (float*)(ws + OFF_B0), 132, 256, HPN, 256);
    prep_pad<<<(HPN * HPK + 255) / 256, 256, 0, stream>>>(
        W1, b1, (__bf16*)(ws + OFF_W1), (float*)(ws + OFF_B1), 132, 132, HPN, HPK);
    prep_pad<<<(HPN * HPK + 255) / 256, 256, 0, stream>>>(
        W2, b2, (__bf16*)(ws + OFF_W2), (float*)(ws + OFF_B2), 132, 132, HPN, HPK);
    prep_pad<<<(DD * HPK + 255) / 256, 256, 0, stream>>>(
        W3, b3, (__bf16*)(ws + OFF_W3), (float*)(ws + OFF_B3), DD, 132, DD, HPK);

    mp_fused<<<BB / 128, 256, SMEM_TOTAL, stream>>>(signal, comps, ws, (float*)d_out);
}
